// TransformerBlock_1_27977416966554
// MI455X (gfx1250) — matrix operand order repaired and checked by reading, not compiled, not run on hardware
//
#include <hip/hip_runtime.h>

// ---------------------------------------------------------------------------
// CDNA5 / gfx1250 transformer block: bf16 WMMA (16x16x32) everywhere, fp32
// softmax/LayerNorm. wave32; every wave owns one 16-row output tile.
// ---------------------------------------------------------------------------

#define DEVINL __device__ __forceinline__

typedef __bf16 bf16;
typedef bf16  bf16x16 __attribute__((ext_vector_type(16)));
typedef float f32x8   __attribute__((ext_vector_type(8)));
typedef unsigned int u32;
typedef u32 u32x8 __attribute__((ext_vector_type(8)));

constexpr int Bb = 8;
constexpr int Ll = 2048;
constexpr int Dd = 64;
constexpr int Hh = 8;
constexpr int LT = Ll / 16;          // 128 row-tiles of 16
constexpr float LN_EPS = 1e-5f;

DEVINL f32x8 wmma_bf16(bf16x16 a, bf16x16 b, f32x8 c) {
  // v_wmma_f32_16x16x32_bf16  D = A(16x32) * B(32x16) + C
  return __builtin_amdgcn_wmma_f32_16x16x32_bf16(false, a, false, b,
                                                 (short)0, c, false, false);
}

DEVINL bf16x16 pack16(uint4 a, uint4 c) {
  u32x8 u;
  u[0]=a.x; u[1]=a.y; u[2]=a.z; u[3]=a.w;
  u[4]=c.x; u[5]=c.y; u[6]=c.z; u[7]=c.w;
  return __builtin_bit_cast(bf16x16, u);
}

// B fragment (K=32 x N=16). Memory: for fixed n, 32 contiguous K elements at
// base + n*stride. Layout matches A: lanes 0-15 hold K=0..7 then 16..23,
// lanes 16-31 hold K=8..15 then 24..31.
DEVINL bf16x16 load_b16(const bf16* base, int stride, int lane) {
  int nn = lane & 15;
  int kb = (lane & 16) >> 1;
  const bf16* r = base + (size_t)nn * stride + kb;
  const uint4* p0 = reinterpret_cast<const uint4*>(r);
  const uint4* p1 = reinterpret_cast<const uint4*>(r + 16);
  return pack16(p0[0], p1[0]);
}

// A fragment (M=16 x K=32) from bf16 row-major rows. ISA layout: lane m=l&15;
// lanes<16: K=0..7 then 16..23; lanes>=16: K=8..15 then 24..31.
DEVINL bf16x16 load_a16(const bf16* base, int stride, int lane) {
  int m  = lane & 15;
  int k0 = (lane & 16) >> 1;          // 0 or 8
  const bf16* r = base + (size_t)m * stride + k0;
  const uint4* p0 = reinterpret_cast<const uint4*>(r);
  const uint4* p1 = reinterpret_cast<const uint4*>(r + 16);
  return pack16(p0[0], p1[0]);
}

// A fragment built from an f32 source (global or LDS), converting to bf16.
DEVINL bf16x16 load_a_f32(const float* base, int stride, int lane) {
  int m  = lane & 15;
  int k0 = (lane & 16) >> 1;
  const float* r = base + (size_t)m * stride + k0;
  bf16x16 v;
  #pragma unroll
  for (int i = 0; i < 8; ++i) v[i]     = (bf16)r[i];
  #pragma unroll
  for (int i = 0; i < 8; ++i) v[i + 8] = (bf16)r[16 + i];
  return v;
}

// Half-wave (16-lane) reductions: C/D fragments put one row across lanes
// 0..15 (or 16..31), so xor masks 1,2,4,8 reduce a row.
DEVINL float hred_max(float x) {
  x = fmaxf(x, __shfl_xor(x, 1, 32));
  x = fmaxf(x, __shfl_xor(x, 2, 32));
  x = fmaxf(x, __shfl_xor(x, 4, 32));
  x = fmaxf(x, __shfl_xor(x, 8, 32));
  return x;
}
DEVINL float hred_sum(float x) {
  x += __shfl_xor(x, 1, 32);
  x += __shfl_xor(x, 2, 32);
  x += __shfl_xor(x, 4, 32);
  x += __shfl_xor(x, 8, 32);
  return x;
}

DEVINL unsigned short bf16bits(float f) {
  return __builtin_bit_cast(unsigned short, (bf16)f);
}

// ---------------------------------------------------------------------------
// K0: weight transpose + bf16 convert. Wt[o][i] = W[i][o] so B-fragments read
// contiguous K (=input dim) per output column n.
// ---------------------------------------------------------------------------
__global__ void prep_weights(const float* Wq, const float* Wk, const float* Wv,
                             const float* We, const float* Wo, const float* w1,
                             const float* w2, bf16* Wqt, bf16* Wkt, bf16* Wvt,
                             bf16* Wet, bf16* Wot, bf16* w1t, bf16* w2t) {
  int i = blockIdx.x * blockDim.x + threadIdx.x;
  const int NP = 512 * 64;
  if (i < 4 * NP) {                               // W_q..W_e: [64,512] -> [512,64]
    int a = i / NP, r = i % NP;
    int o = r / 64, k = r % 64;
    const float* W = (a == 0) ? Wq : (a == 1) ? Wk : (a == 2) ? Wv : We;
    bf16* T = (a == 0) ? Wqt : (a == 1) ? Wkt : (a == 2) ? Wvt : Wet;
    T[o * 64 + k] = (bf16)W[k * 512 + o];
  } else if (i < 5 * NP) {                        // W_o: [512,64] -> [64,512]
    int r = i - 4 * NP;
    int o = r / 512, k = r % 512;
    Wot[o * 512 + k] = (bf16)Wo[k * 64 + o];
  } else if (i < 5 * NP + 2 * 4096) {             // w1,w2: [64,64] -> [64,64]^T
    int r = i - 5 * NP;
    int a = r / 4096; r %= 4096;
    int o = r / 64, k = r % 64;
    (a ? w2t : w1t)[o * 64 + k] = (bf16)((a ? w2 : w1)[k * 64 + o]);
  }
}

// ---------------------------------------------------------------------------
// K1: head projections. One wave per (proj, b, h, 16-row tile).
// Qh,Kh: [B,H,L,64] bf16 (d-contiguous). VEt = (V@Wv + E@We) stored
// transposed [B,H,64,L] bf16 (l-contiguous) for P@VE B-fragments.
// ---------------------------------------------------------------------------
__global__ __launch_bounds__(128) void proj_kernel(
    const float* Q, const float* K, const float* V, const float* E,
    const bf16* Wqt, const bf16* Wkt, const bf16* Wvt, const bf16* Wet,
    bf16* Qh, bf16* Kh, bf16* VEt) {
  int wid  = blockIdx.x * 4 + (threadIdx.x >> 5);
  int lane = threadIdx.x & 31;
  int p    = wid / (Bb * Hh * LT);
  int rem  = wid % (Bb * Hh * LT);
  int b    = rem / (Hh * LT); rem %= (Hh * LT);
  int h    = rem / LT;
  int q0   = (rem % LT) * 16;

  f32x8 acc[4] = {};
  auto mm = [&](const float* X, const bf16* Wt) {
    #pragma unroll
    for (int s = 0; s < 2; ++s) {
      bf16x16 a = load_a_f32(X + ((size_t)b * Ll + q0) * 64 + 32 * s, 64, lane);
      #pragma unroll
      for (int t = 0; t < 4; ++t) {
        bf16x16 bb = load_b16(Wt + (h * 64 + t * 16) * 64 + 32 * s, 64, lane);
        acc[t] = wmma_bf16(a, bb, acc[t]);
      }
    }
  };
  if (p == 0)      mm(Q, Wqt);
  else if (p == 1) mm(K, Wkt);
  else           { mm(V, Wvt); mm(E, Wet); }

  int nn = lane & 15, hi8 = (lane >> 4) * 8;
  size_t bh = (size_t)b * Hh + h;
  if (p < 2) {
    bf16* O = (p == 0) ? Qh : Kh;
    #pragma unroll
    for (int t = 0; t < 4; ++t)
      #pragma unroll
      for (int v = 0; v < 8; ++v)
        O[(bh * Ll + q0 + v + hi8) * 64 + t * 16 + nn] = (bf16)acc[t][v];
  } else {
    #pragma unroll
    for (int t = 0; t < 4; ++t)
      #pragma unroll
      for (int v = 0; v < 8; ++v)
        VEt[(bh * 64 + t * 16 + nn) * Ll + q0 + v + hi8] = (bf16)acc[t][v];
  }
}

// ---------------------------------------------------------------------------
// K2: causal flash attention. One wave per (b,h,16-row q tile); KV tiles of
// 32 so P(16x32) is exactly one bf16 A-fragment. fp32 online softmax.
// ---------------------------------------------------------------------------
__global__ __launch_bounds__(128) void attn_kernel(
    const bf16* Qh, const bf16* Kh, const bf16* VEt,
    const unsigned char* maskp, bf16* att_out) {
  __shared__ unsigned short plds[4][16 * 32];   // per-wave P tile (bf16 bits)
  int wslot = threadIdx.x >> 5;
  int wid   = blockIdx.x * 4 + wslot;
  int lane  = threadIdx.x & 31;
  int b     = wid / (Hh * LT);
  int rem   = wid % (Hh * LT);
  int h     = rem / LT;
  int q0    = (rem % LT) * 16;
  size_t bh = (size_t)b * Hh + h;
  int nn = lane & 15, hi8 = (lane >> 4) * 8;

  bf16x16 qa0 = load_a16(Qh + (bh * Ll + q0) * 64,      64, lane);
  bf16x16 qa1 = load_a16(Qh + (bh * Ll + q0) * 64 + 32, 64, lane);

  float rm[8], rl[8];
  bool pmask[8];
  f32x8 o0 = {}, o1 = {}, o2 = {}, o3 = {};
  #pragma unroll
  for (int v = 0; v < 8; ++v) {
    rm[v] = -1e30f; rl[v] = 0.f;
    pmask[v] = maskp[(size_t)b * Ll + q0 + v + hi8] != 0;
  }

  const float scale = 1.0f / (8.0f + 1e-6f);   // 1/(sqrt(64)+EPS)
  int jt = (q0 + 15) / 32 + 1;                 // causal tile count
  unsigned short* pl = plds[wslot];

  for (int jj = 0; jj < jt; ++jj) {
    int j = jj * 32;
    f32x8 s0 = {}, s1 = {};
    {
      bf16x16 b00 = load_b16(Kh + (bh * Ll + j)      * 64,      64, lane);
      bf16x16 b10 = load_b16(Kh + (bh * Ll + j + 16) * 64,      64, lane);
      s0 = wmma_bf16(qa0, b00, s0);
      s1 = wmma_bf16(qa0, b10, s1);
      bf16x16 b01 = load_b16(Kh + (bh * Ll + j)      * 64 + 32, 64, lane);
      bf16x16 b11 = load_b16(Kh + (bh * Ll + j + 16) * 64 + 32, 64, lane);
      s0 = wmma_bf16(qa1, b01, s0);
      s1 = wmma_bf16(qa1, b11, s1);
    }
    #pragma unroll
    for (int v = 0; v < 8; ++v) {
      int q = q0 + v + hi8;
      float a0 = s0[v] * scale; if ((j + nn) > q      || pmask[v]) a0 = -100.f;
      float a1 = s1[v] * scale; if ((j + 16 + nn) > q || pmask[v]) a1 = -100.f;
      float mt = hred_max(fmaxf(a0, a1));
      float mn = fmaxf(rm[v], mt);
      float corr = __expf(rm[v] - mn);
      rm[v] = mn;
      float p0 = __expf(a0 - mn), p1 = __expf(a1 - mn);
      rl[v] = rl[v] * corr + hred_sum(p0 + p1);
      o0[v] *= corr; o1[v] *= corr; o2[v] *= corr; o3[v] *= corr;
      int row = v + hi8;
      pl[row * 32 + nn]      = bf16bits(p0);
      pl[row * 32 + 16 + nn] = bf16bits(p1);
    }
    asm volatile("s_wait_dscnt 0x0" ::: "memory");   // ds stores -> ds loads
    bf16x16 pa = load_a16((const bf16*)pl, 32, lane);
    o0 = wmma_bf16(pa, load_b16(VEt + (bh * 64 +  0) * Ll + j, Ll, lane), o0);
    o1 = wmma_bf16(pa, load_b16(VEt + (bh * 64 + 16) * Ll + j, Ll, lane), o1);
    o2 = wmma_bf16(pa, load_b16(VEt + (bh * 64 + 32) * Ll + j, Ll, lane), o2);
    o3 = wmma_bf16(pa, load_b16(VEt + (bh * 64 + 48) * Ll + j, Ll, lane), o3);
  }

  #pragma unroll
  for (int v = 0; v < 8; ++v) {
    float inv = 1.0f / rl[v];
    int row = q0 + v + hi8;
    bf16* dst = att_out + ((size_t)b * Ll + row) * 512 + h * 64;
    dst[ 0 + nn] = (bf16)(o0[v] * inv);
    dst[16 + nn] = (bf16)(o1[v] * inv);
    dst[32 + nn] = (bf16)(o2[v] * inv);
    dst[48 + nn] = (bf16)(o3[v] * inv);
  }
}

// ---------------------------------------------------------------------------
// K3: W_o projection + residual + LN + FFN(relu) + residual + LN, fused.
// One wave per (b, 16-row tile). X and H1 staged in LDS for A-fragments.
// ---------------------------------------------------------------------------
__global__ __launch_bounds__(128) void out_kernel(
    const bf16* att_out, const bf16* Wot, const bf16* w1t, const bf16* w2t,
    const float* Vin, const float* b1, const float* b2,
    const float* lng, const float* lnb, float* out) {
  __shared__ float xls[4][16 * 64];
  __shared__ float hls[4][16 * 64];
  int wslot = threadIdx.x >> 5;
  int wid   = blockIdx.x * 4 + wslot;
  int lane  = threadIdx.x & 31;
  int b     = wid / LT;
  int q0    = (wid % LT) * 16;
  int nn = lane & 15, hi8 = (lane >> 4) * 8;
  float* xl = xls[wslot];
  float* hl = hls[wslot];

  // V_att = att_out[16,512] @ W_o[512,64]
  f32x8 acc[4] = {};
  const bf16* arow = att_out + ((size_t)b * Ll + q0) * 512;
  for (int s = 0; s < 16; ++s) {
    bf16x16 a = load_a16(arow + 32 * s, 512, lane);
    #pragma unroll
    for (int t = 0; t < 4; ++t)
      acc[t] = wmma_bf16(a, load_b16(Wot + (t * 16) * 512 + 32 * s, 512, lane),
                         acc[t]);
  }

  float g[4], be[4], bb1[4], bb2[4];
  #pragma unroll
  for (int t = 0; t < 4; ++t) {
    g[t] = lng[t * 16 + nn]; be[t] = lnb[t * 16 + nn];
    bb1[t] = b1[t * 16 + nn]; bb2[t] = b2[t * 16 + nn];
  }

  // X = LN(V + V_att) -> LDS (fp32)
  #pragma unroll
  for (int v = 0; v < 8; ++v) {
    int row = v + hi8;
    const float* vr = Vin + ((size_t)b * Ll + q0 + row) * 64;
    float x0 = acc[0][v] + vr[ 0 + nn];
    float x1 = acc[1][v] + vr[16 + nn];
    float x2 = acc[2][v] + vr[32 + nn];
    float x3 = acc[3][v] + vr[48 + nn];
    float s1 = hred_sum(x0 + x1 + x2 + x3);
    float s2 = hred_sum(x0*x0 + x1*x1 + x2*x2 + x3*x3);
    float mu = s1 * (1.f / 64.f);
    float var = s2 * (1.f / 64.f) - mu * mu;
    float rs = rsqrtf(var + LN_EPS);
    xl[row * 64 +  0 + nn] = (x0 - mu) * rs * g[0] + be[0];
    xl[row * 64 + 16 + nn] = (x1 - mu) * rs * g[1] + be[1];
    xl[row * 64 + 32 + nn] = (x2 - mu) * rs * g[2] + be[2];
    xl[row * 64 + 48 + nn] = (x3 - mu) * rs * g[3] + be[3];
  }
  asm volatile("s_wait_dscnt 0x0" ::: "memory");

  // H1 = relu(X @ w1 + b1) -> LDS
  f32x8 hacc[4] = {};
  #pragma unroll
  for (int s = 0; s < 2; ++s) {
    bf16x16 a = load_a_f32(xl + 32 * s, 64, lane);
    #pragma unroll
    for (int t = 0; t < 4; ++t)
      hacc[t] = wmma_bf16(a, load_b16(w1t + (t * 16) * 64 + 32 * s, 64, lane),
                          hacc[t]);
  }
  #pragma unroll
  for (int v = 0; v < 8; ++v) {
    int row = v + hi8;
    #pragma unroll
    for (int t = 0; t < 4; ++t)
      hl[row * 64 + t * 16 + nn] = fmaxf(hacc[t][v] + bb1[t], 0.f);
  }
  asm volatile("s_wait_dscnt 0x0" ::: "memory");

  // Y = H1 @ w2 + b2 + X ; out = LN(Y)
  f32x8 yacc[4] = {};
  #pragma unroll
  for (int s = 0; s < 2; ++s) {
    bf16x16 a = load_a_f32(hl + 32 * s, 64, lane);
    #pragma unroll
    for (int t = 0; t < 4; ++t)
      yacc[t] = wmma_bf16(a, load_b16(w2t + (t * 16) * 64 + 32 * s, 64, lane),
                          yacc[t]);
  }
  #pragma unroll
  for (int v = 0; v < 8; ++v) {
    int row = v + hi8;
    float y0 = yacc[0][v] + bb2[0] + xl[row * 64 +  0 + nn];
    float y1 = yacc[1][v] + bb2[1] + xl[row * 64 + 16 + nn];
    float y2 = yacc[2][v] + bb2[2] + xl[row * 64 + 32 + nn];
    float y3 = yacc[3][v] + bb2[3] + xl[row * 64 + 48 + nn];
    float s1 = hred_sum(y0 + y1 + y2 + y3);
    float s2 = hred_sum(y0*y0 + y1*y1 + y2*y2 + y3*y3);
    float mu = s1 * (1.f / 64.f);
    float var = s2 * (1.f / 64.f) - mu * mu;
    float rs = rsqrtf(var + LN_EPS);
    float* orow = out + ((size_t)b * Ll + q0 + row) * 64;
    orow[ 0 + nn] = (y0 - mu) * rs * g[0] + be[0];
    orow[16 + nn] = (y1 - mu) * rs * g[1] + be[1];
    orow[32 + nn] = (y2 - mu) * rs * g[2] + be[2];
    orow[48 + nn] = (y3 - mu) * rs * g[3] + be[3];
  }
}

// ---------------------------------------------------------------------------
extern "C" void kernel_launch(void* const* d_in, const int* in_sizes, int n_in,
                              void* d_out, int out_size, void* d_ws, size_t ws_size,
                              hipStream_t stream) {
  const float* Q   = (const float*)d_in[0];
  const float* K   = (const float*)d_in[1];
  const float* V   = (const float*)d_in[2];
  const float* E   = (const float*)d_in[3];
  const float* Wq  = (const float*)d_in[4];
  const float* Wk  = (const float*)d_in[5];
  const float* Wv  = (const float*)d_in[6];
  const float* We  = (const float*)d_in[7];
  const float* Wo  = (const float*)d_in[8];
  const float* w1  = (const float*)d_in[9];
  const float* b1  = (const float*)d_in[10];
  const float* w2  = (const float*)d_in[11];
  const float* b2  = (const float*)d_in[12];
  const float* lng = (const float*)d_in[13];
  const float* lnb = (const float*)d_in[14];
  const unsigned char* mask = (const unsigned char*)d_in[15];
  float* out = (float*)d_out;

  char* ws = (char*)d_ws;
  size_t off = 0;
  auto carve = [&](size_t bytes) -> char* {
    char* p = ws + off;
    off += (bytes + 255) & ~(size_t)255;
    return p;
  };
  const size_t BHLD = (size_t)Bb * Hh * Ll * Dd;      // 8.39M elements
  bf16* Qh   = (bf16*)carve(BHLD * 2);                // [B,H,L,64]
  bf16* Kh   = (bf16*)carve(BHLD * 2);                // [B,H,L,64]
  bf16* VEt  = (bf16*)carve(BHLD * 2);                // [B,H,64,L]
  bf16* attO = (bf16*)carve((size_t)Bb * Ll * 512 * 2); // [B*L,512]
  bf16* Wqt  = (bf16*)carve(512 * 64 * 2);
  bf16* Wkt  = (bf16*)carve(512 * 64 * 2);
  bf16* Wvt  = (bf16*)carve(512 * 64 * 2);
  bf16* Wet  = (bf16*)carve(512 * 64 * 2);
  bf16* Wot  = (bf16*)carve(64 * 512 * 2);
  bf16* w1t  = (bf16*)carve(64 * 64 * 2);
  bf16* w2t  = (bf16*)carve(64 * 64 * 2);

  // K0: weight prep (172032 elements)
  prep_weights<<<672, 256, 0, stream>>>(Wq, Wk, Wv, We, Wo, w1, w2,
                                        Wqt, Wkt, Wvt, Wet, Wot, w1t, w2t);
  // K1: 3 proj types * B*H*(L/16) tiles = 24576 waves
  proj_kernel<<<6144, 128, 0, stream>>>(Q, K, V, E, Wqt, Wkt, Wvt, Wet,
                                        Qh, Kh, VEt);
  // K2: B*H*(L/16) = 8192 waves
  attn_kernel<<<2048, 128, 0, stream>>>(Qh, Kh, VEt, mask, attO);
  // K3: B*(L/16) = 1024 waves
  out_kernel<<<256, 128, 0, stream>>>(attO, Wot, w1t, w2t, V, b1, b2,
                                      lng, lnb, out);
}
